// PyTorchLSTM_71219147702440
// MI455X (gfx1250) — matrix operand order repaired and checked by reading, not compiled, not run on hardware
//
#include <hip/hip_runtime.h>
#include <stdint.h>

typedef __attribute__((ext_vector_type(16))) __bf16 v16bf;
typedef __attribute__((ext_vector_type(8)))  float  v8f;

#define B_   64
#define T_   512
#define IN_  512
#define H_   1024
#define G4_  4096
#define A_   512
#define BT_  (B_ * T_)

// ---------------- helpers ----------------

__device__ __forceinline__ uint16_t f2bf(float f) {
  uint32_t u = __float_as_uint(f);
  uint32_t r = u + 0x7FFFu + ((u >> 16) & 1u);   // round-to-nearest-even
  return (uint16_t)(r >> 16);
}

// A-matrix fragment (16x32 bf16) per CDNA5 ISA layout, from bf16 row-major [., ldk]:
// lane l: M = l&15 ; elems j=0..7 -> K = hi*8+j ; j=8..15 -> K = 16+hi*8+(j-8)
__device__ __forceinline__ v16bf load_a_frag(const uint16_t* __restrict__ Abase, int ldk, int k0) {
  int lane = threadIdx.x & 31;
  int m    = lane & 15;
  int khi  = (lane >> 4) << 3;
  const uint16_t* p = Abase + (size_t)m * ldk + k0 + khi;
  union { v16bf v; uint4 u[2]; } r;
  r.u[0] = *(const uint4*)(p);        // K = khi .. khi+7
  r.u[1] = *(const uint4*)(p + 16);   // K = 16+khi .. 16+khi+7
  return r.v;
}

// B-matrix fragment (32x16 bf16) per CDNA5 ISA layout, from bf16 row-major W[N, ldk]:
// lane l: N = l&15 ; elems j=0..7 -> K = hi*8+j ; j=8..15 -> K = 16+hi*8+(j-8)
__device__ __forceinline__ v16bf load_b_frag(const uint16_t* __restrict__ Wbase, int ldk, int k0) {
  int lane = threadIdx.x & 31;
  int n    = lane & 15;
  int kof  = (lane >> 4) << 3;
  const uint16_t* p = Wbase + (size_t)n * ldk + k0 + kof;
  union { v16bf v; uint4 u[2]; } r;
  r.u[0] = *(const uint4*)(p);
  r.u[1] = *(const uint4*)(p + 16);
  return r.v;
}

// branchless fast activations (v_exp + v_rcp, no EXEC games)
__device__ __forceinline__ float sigm(float x) {
  return __builtin_amdgcn_rcpf(1.0f + __expf(-x));
}
__device__ __forceinline__ float tanh_fast(float x) {
  float e = __expf(2.0f * x);
  return 1.0f - 2.0f * __builtin_amdgcn_rcpf(e + 1.0f);  // saturates to +/-1
}

#define WMMA_BF16(Acc, Af, Bf) \
  __builtin_amdgcn_wmma_f32_16x16x32_bf16(false, (Af), false, (Bf), (short)0, (Acc), false, false)

// ---------------- prep kernels ----------------

__global__ void cvt_f32_to_bf16(const float* __restrict__ src, uint16_t* __restrict__ dst, int n) {
  int i = blockIdx.x * blockDim.x + threadIdx.x;
  if (i < n) dst[i] = f2bf(src[i]);
}

__global__ void add_bias_vec(const float* __restrict__ a, const float* __restrict__ b,
                             float* __restrict__ out, int n) {
  int i = blockIdx.x * blockDim.x + threadIdx.x;
  if (i < n) out[i] = a[i] + b[i];
}

// ---------------- generic GEMM: C[M,N] = act(A[M,K] * W[N,K]^T + bias) ----------------
// A bf16 row-major, W bf16 row-major [N,K]; one wave -> 16x64 strip.
// Software-pipelined K loop: next-iteration fragments are loaded before the
// current WMMAs so VMEM latency overlaps matrix ops within one wave.

template <int ACT, int OBF>  // ACT: 0 none, 1 tanh ; OBF: 1 bf16 out, 0 f32 out
__global__ void __launch_bounds__(256)
gemm_bf16_nt4(const uint16_t* __restrict__ A, const uint16_t* __restrict__ W,
              const float* __restrict__ bias, void* __restrict__ Cv,
              int M, int N, int K) {
  int lane = threadIdx.x & 31;
  int wave = blockIdx.x * (blockDim.x >> 5) + (threadIdx.x >> 5);
  int ngroups = N >> 6;
  int mt = wave / ngroups;
  int ng = wave - mt * ngroups;
  if ((mt << 4) >= M) return;           // uniform per wave
  int n0 = ng << 6;

  v8f acc0 = {}, acc1 = {}, acc2 = {}, acc3 = {};
  const uint16_t* Abase = A + ((size_t)mt << 4) * (size_t)K;
  const uint16_t* W0 = W + (size_t)(n0 +  0) * (size_t)K;
  const uint16_t* W1 = W + (size_t)(n0 + 16) * (size_t)K;
  const uint16_t* W2 = W + (size_t)(n0 + 32) * (size_t)K;
  const uint16_t* W3 = W + (size_t)(n0 + 48) * (size_t)K;

  // prologue
  v16bf a  = load_a_frag(Abase, K, 0);
  v16bf b0 = load_b_frag(W0, K, 0);
  v16bf b1 = load_b_frag(W1, K, 0);
  v16bf b2 = load_b_frag(W2, K, 0);
  v16bf b3 = load_b_frag(W3, K, 0);

  for (int k0 = 32; k0 < K; k0 += 32) {
    v16bf an  = load_a_frag(Abase, K, k0);
    v16bf bn0 = load_b_frag(W0, K, k0);
    v16bf bn1 = load_b_frag(W1, K, k0);
    v16bf bn2 = load_b_frag(W2, K, k0);
    v16bf bn3 = load_b_frag(W3, K, k0);
    acc0 = WMMA_BF16(acc0, a, b0);
    acc1 = WMMA_BF16(acc1, a, b1);
    acc2 = WMMA_BF16(acc2, a, b2);
    acc3 = WMMA_BF16(acc3, a, b3);
    a = an; b0 = bn0; b1 = bn1; b2 = bn2; b3 = bn3;
  }
  acc0 = WMMA_BF16(acc0, a, b0);
  acc1 = WMMA_BF16(acc1, a, b1);
  acc2 = WMMA_BF16(acc2, a, b2);
  acc3 = WMMA_BF16(acc3, a, b3);

  int n = lane & 15, hi = lane >> 4;
#pragma unroll
  for (int q = 0; q < 4; ++q) {
    v8f acc = (q == 0) ? acc0 : (q == 1) ? acc1 : (q == 2) ? acc2 : acc3;
    float bv = bias[n0 + (q << 4) + n];
#pragma unroll
    for (int v = 0; v < 8; ++v) {
      float val = acc[v] + bv;
      if (ACT == 1) val = tanh_fast(val);
      size_t idx = (size_t)((mt << 4) + v + (hi << 3)) * (size_t)N
                 + (size_t)(n0 + (q << 4) + n);
      if (OBF) ((uint16_t*)Cv)[idx] = f2bf(val);
      else     ((float*)Cv)[idx]    = val;
    }
  }
}

// ---------------- LSTM step ----------------
// gates[B,4H] = gx[:,t,:] + h_in * w_hh^T ; fused elementwise LSTM update.
// 256 waves: mt 0..3 (batch tile), jt 0..63 (hidden column tile).
// gx and c prefetched before the K loop so their latency hides under the GEMM.

__global__ void __launch_bounds__(128)
lstm_step(const float* __restrict__ gx, const uint16_t* __restrict__ whh,
          const uint16_t* __restrict__ h_in, uint16_t* __restrict__ h_out,
          float* __restrict__ c, uint16_t* __restrict__ hs, int t) {
  int lane = threadIdx.x & 31;
  int wave = blockIdx.x * (blockDim.x >> 5) + (threadIdx.x >> 5);  // 0..255
  int mt = wave >> 6;   // 0..3
  int jt = wave & 63;   // 0..63

  int n = lane & 15, hi = lane >> 4;
  int j = (jt << 4) + n;

  // prefetch gx (4 gates x 8 rows) and c (8 rows) — issued before the GEMM loads
  float gxi[8], gxf[8], gxg[8], gxo[8], cold[8];
#pragma unroll
  for (int v = 0; v < 8; ++v) {
    int b = (mt << 4) + v + (hi << 3);
    size_t grow = ((size_t)b * T_ + (size_t)t) * (size_t)G4_;
    gxi[v] = gx[grow + 0 * H_ + j];
    gxf[v] = gx[grow + 1 * H_ + j];
    gxg[v] = gx[grow + 2 * H_ + j];
    gxo[v] = gx[grow + 3 * H_ + j];
    cold[v] = c[(size_t)b * H_ + (size_t)j];
  }

  v8f acc0 = {}, acc1 = {}, acc2 = {}, acc3 = {};
  const uint16_t* Abase = h_in + ((size_t)mt << 4) * (size_t)H_;
  const uint16_t* W0 = whh + (size_t)(0 * H_ + (jt << 4)) * (size_t)H_;
  const uint16_t* W1 = whh + (size_t)(1 * H_ + (jt << 4)) * (size_t)H_;
  const uint16_t* W2 = whh + (size_t)(2 * H_ + (jt << 4)) * (size_t)H_;
  const uint16_t* W3 = whh + (size_t)(3 * H_ + (jt << 4)) * (size_t)H_;

  v16bf a  = load_a_frag(Abase, H_, 0);
  v16bf b0 = load_b_frag(W0, H_, 0);
  v16bf b1 = load_b_frag(W1, H_, 0);
  v16bf b2 = load_b_frag(W2, H_, 0);
  v16bf b3 = load_b_frag(W3, H_, 0);

  for (int k0 = 32; k0 < H_; k0 += 32) {
    v16bf an  = load_a_frag(Abase, H_, k0);
    v16bf bn0 = load_b_frag(W0, H_, k0);
    v16bf bn1 = load_b_frag(W1, H_, k0);
    v16bf bn2 = load_b_frag(W2, H_, k0);
    v16bf bn3 = load_b_frag(W3, H_, k0);
    acc0 = WMMA_BF16(acc0, a, b0);
    acc1 = WMMA_BF16(acc1, a, b1);
    acc2 = WMMA_BF16(acc2, a, b2);
    acc3 = WMMA_BF16(acc3, a, b3);
    a = an; b0 = bn0; b1 = bn1; b2 = bn2; b3 = bn3;
  }
  acc0 = WMMA_BF16(acc0, a, b0);
  acc1 = WMMA_BF16(acc1, a, b1);
  acc2 = WMMA_BF16(acc2, a, b2);
  acc3 = WMMA_BF16(acc3, a, b3);

#pragma unroll
  for (int v = 0; v < 8; ++v) {
    int b = (mt << 4) + v + (hi << 3);
    float i_ = sigm(acc0[v] + gxi[v]);
    float f_ = sigm(acc1[v] + gxf[v]);
    float g_ = tanh_fast(acc2[v] + gxg[v]);
    float o_ = sigm(acc3[v] + gxo[v]);
    float cn = f_ * cold[v] + i_ * g_;
    size_t cix = (size_t)b * H_ + (size_t)j;
    c[cix] = cn;
    uint16_t hn = f2bf(o_ * tanh_fast(cn));
    h_out[cix] = hn;
    hs[((size_t)b * T_ + (size_t)t) * (size_t)H_ + (size_t)j] = hn;
  }
}

// ---------------- launch ----------------

extern "C" void kernel_launch(void* const* d_in, const int* in_sizes, int n_in,
                              void* d_out, int out_size, void* d_ws, size_t ws_size,
                              hipStream_t stream) {
  const float* x   = (const float*)d_in[0];   // [64,512,512]
  const float* w1  = (const float*)d_in[1];   // [1024,512]
  const float* b1  = (const float*)d_in[2];   // [1024]
  const float* wih = (const float*)d_in[3];   // [4096,1024]
  const float* whh = (const float*)d_in[4];   // [4096,1024]
  const float* bih = (const float*)d_in[5];   // [4096]
  const float* bhh = (const float*)d_in[6];   // [4096]
  const float* w3  = (const float*)d_in[7];   // [512,1024]
  const float* b3  = (const float*)d_in[8];   // [512]
  float* out = (float*)d_out;                 // [64,512,512]

  char* ws = (char*)d_ws;
  auto carve = [&](size_t bytes) -> char* {
    char* p = ws;
    ws += (bytes + 255) & ~(size_t)255;
    return p;
  };
  uint16_t* w1b   = (uint16_t*)carve((size_t)H_  * IN_ * 2);
  uint16_t* wihb  = (uint16_t*)carve((size_t)G4_ * H_  * 2);
  uint16_t* whhb  = (uint16_t*)carve((size_t)G4_ * H_  * 2);
  uint16_t* w3b   = (uint16_t*)carve((size_t)A_  * H_  * 2);
  float*    bias12= (float*)   carve((size_t)G4_ * 4);
  uint16_t* xb    = (uint16_t*)carve((size_t)BT_ * IN_ * 2);
  uint16_t* fbuf  = (uint16_t*)carve((size_t)BT_ * H_  * 2);
  float*    gxbuf = (float*)   carve((size_t)BT_ * G4_ * 4);
  uint16_t* hsbuf = (uint16_t*)carve((size_t)BT_ * H_  * 2);
  uint16_t* h0    = (uint16_t*)carve((size_t)B_  * H_  * 2);
  uint16_t* h1    = (uint16_t*)carve((size_t)B_  * H_  * 2);
  float*    cbuf  = (float*)   carve((size_t)B_  * H_  * 4);

  // 1) convert weights + x to bf16, fuse recurrent biases
  cvt_f32_to_bf16<<<(H_ * IN_ + 255) / 256, 256, 0, stream>>>(w1,  w1b,  H_ * IN_);
  cvt_f32_to_bf16<<<(G4_ * H_ + 255) / 256, 256, 0, stream>>>(wih, wihb, G4_ * H_);
  cvt_f32_to_bf16<<<(G4_ * H_ + 255) / 256, 256, 0, stream>>>(whh, whhb, G4_ * H_);
  cvt_f32_to_bf16<<<(A_ * H_ + 255) / 256, 256, 0, stream>>>(w3,  w3b,  A_ * H_);
  cvt_f32_to_bf16<<<(BT_ * IN_ + 255) / 256, 256, 0, stream>>>(x, xb, BT_ * IN_);
  add_bias_vec<<<(G4_ + 255) / 256, 256, 0, stream>>>(bih, bhh, bias12, G4_);

  // 2) zero initial state (capture-legal stream ops)
  hipMemsetAsync(h0,   0, (size_t)B_ * H_ * 2, stream);
  hipMemsetAsync(cbuf, 0, (size_t)B_ * H_ * 4, stream);

  // 3) fc1 + tanh : f = tanh(x * w1^T + b1)      M=32768 N=1024 K=512  (bf16 out)
  {
    int waves = (BT_ / 16) * (H_ / 64);   // 32768
    gemm_bf16_nt4<1, 1><<<waves / 8, 256, 0, stream>>>(xb, w1b, b1, fbuf, BT_, H_, IN_);
  }
  // 4) gx = f * w_ih^T + (b_ih + b_hh)           M=32768 N=4096 K=1024 (f32 out)
  {
    int waves = (BT_ / 16) * (G4_ / 64);  // 131072
    gemm_bf16_nt4<0, 0><<<waves / 8, 256, 0, stream>>>(fbuf, wihb, bias12, gxbuf, BT_, G4_, H_);
  }
  // 5) sequential LSTM scan, one launch per timestep (w_hh bf16 = 8MB stays in 192MB L2)
  for (int t = 0; t < T_; ++t) {
    const uint16_t* hin = (t & 1) ? h1 : h0;
    uint16_t*       hout = (t & 1) ? h0 : h1;
    lstm_step<<<64, 128, 0, stream>>>(gxbuf, whhb, hin, hout, cbuf, hsbuf, t);
  }
  // 6) fc3 : out = hs * w3^T + b3                M=32768 N=512 K=1024  (f32 out)
  {
    int waves = (BT_ / 16) * (A_ / 64);   // 16384
    gemm_bf16_nt4<0, 0><<<waves / 8, 256, 0, stream>>>(hsbuf, w3b, b3, out, BT_, A_, H_);
  }
}
